// LDAHead_26371099197590
// MI455X (gfx1250) — matrix operand order repaired and checked by reading, not compiled, not run on hardware
//
#include <hip/hip_runtime.h>

typedef __attribute__((ext_vector_type(16))) __bf16 v16bf;
typedef __attribute__((ext_vector_type(8)))  float  v8f;
typedef __attribute__((ext_vector_type(4)))  float  f4;

#define NC 64   // number of components C
#define ND 64   // feature dim D

__device__ __forceinline__ float sq4(f4 a) {
    // 4-element sum of squares, depth 3, independent of other partials
    return (a.x * a.x + a.y * a.y) + (a.z * a.z + a.w * a.w);
}

// out[n,c] = bias[c] + inv_var * dot(z[n], mu[c]) - 0.5*inv_var*||z[n]||^2
// bias[c]  = log_softmax(prior)[c] - 0.5*inv_var*||mu[c]||^2 - 0.5*D*log_cov
__global__ __launch_bounds__(256) void lda_head_kernel(
    const float* __restrict__ z, const float* __restrict__ mu,
    const float* __restrict__ log_cov, const float* __restrict__ prior,
    float* __restrict__ out, int n_rows)
{
    __shared__ float s_bias[NC];

    const int tid = threadIdx.x;
    const float lc = log_cov[0];
    const float inv_var = __expf(-lc);

    // Per-block tiny precompute of the per-class bias (redundant across blocks; negligible).
    if (tid < NC) {
        float mx = -3.4e38f;
        #pragma unroll
        for (int c2 = 0; c2 < NC; ++c2) mx = fmaxf(mx, prior[c2]);
        float s = 0.f;
        #pragma unroll
        for (int c2 = 0; c2 < NC; ++c2) s += __expf(prior[c2] - mx);
        const float lse = mx + __logf(s);
        float msq = 0.f;
        const float* mr = mu + tid * ND;
        #pragma unroll
        for (int d = 0; d < ND; ++d) msq += mr[d] * mr[d];
        s_bias[tid] = (prior[tid] - lse) - 0.5f * inv_var * msq - 0.5f * (float)ND * lc;
    }
    __syncthreads();

    const int lane = tid & 31;
    const int l15  = lane & 15;
    const int hs   = lane >> 4;      // which K-half this lane holds

    // ---- B fragments: mu^T, loaded once per wave, kept in VGPRs ----
    // B as packed here (KxN=32x16, bf16): lane n = column (class index), lanes 0-15 hold K=0..15,
    // lanes 16-31 hold K=16..31, packed sequentially in pairs across 8 VGPRs (contiguous order chosen here, matching A).
    // B[k][c] = mu[c][k], so lane c loads 16 consecutive floats of mu row c.
    v16bf bfrag[4][2];
    #pragma unroll
    for (int ct = 0; ct < 4; ++ct) {
        const float* mrow = mu + (ct * 16 + l15) * ND;
        #pragma unroll
        for (int kh = 0; kh < 2; ++kh) {
            const float* p = mrow + kh * 32 + hs * 16;
            #pragma unroll
            for (int i = 0; i < 16; ++i) bfrag[ct][kh][i] = (__bf16)p[i];
        }
    }

    // Per-lane class bias for the epilogue (column = ct*16 + l15).
    float bias_c[4];
    #pragma unroll
    for (int ct = 0; ct < 4; ++ct) bias_c[ct] = s_bias[ct * 16 + l15];

    const int wave   = blockIdx.x * (blockDim.x >> 5) + (tid >> 5);
    const int nwaves = gridDim.x * (blockDim.x >> 5);
    const int ntiles = n_rows >> 4;

    // Pointer-increment grid-stride loop: avoids 64-bit addr recompute per tile.
    const size_t zstep = (size_t)nwaves * 16 * ND;   // floats per stride
    const size_t ostep = (size_t)nwaves * 16 * NC;
    const float* zp = z   + (size_t)(wave * 16 + l15) * ND;   // this lane's row
    float*       op = out + (size_t)(wave * 16) * NC;         // tile base

    for (int tile = wave; tile < ntiles; tile += nwaves, zp += zstep, op += ostep) {
        // Speculative prefetch of next tile's z rows (2x128B lines per row,
        // split across the two K-half lanes).
        __builtin_prefetch(zp + zstep + hs * 32, 0, 0);

        // ---- A fragments (z tile, 16x64 split into two 16x32 bf16 frags) ----
        // A fragment as packed here (16x32): lane = row (M=l15); lanes hold
        // K = hs*16 + 0..15 packed sequentially (contiguous order, same as B; not AMD's table order).
        v16bf afrag[2];
        float part[8];                 // independent sum-of-squares partials
        #pragma unroll
        for (int kh = 0; kh < 2; ++kh) {
            const f4* p0 = (const f4*)(zp + kh * 32 + hs * 16);
            const f4* p1 = (const f4*)(zp + kh * 32 + hs * 16 + 8);
            f4 f0 = __builtin_nontemporal_load(p0);
            f4 f1 = __builtin_nontemporal_load(p0 + 1);
            f4 f2 = __builtin_nontemporal_load(p1);
            f4 f3 = __builtin_nontemporal_load(p1 + 1);
            part[kh * 4 + 0] = sq4(f0);
            part[kh * 4 + 1] = sq4(f1);
            part[kh * 4 + 2] = sq4(f2);
            part[kh * 4 + 3] = sq4(f3);
            float v[16] = {f0.x,f0.y,f0.z,f0.w, f1.x,f1.y,f1.z,f1.w,
                           f2.x,f2.y,f2.z,f2.w, f3.x,f3.y,f3.z,f3.w};
            #pragma unroll
            for (int i = 0; i < 16; ++i) afrag[kh][i] = (__bf16)v[i];
        }
        // Tree combine: depth 3 instead of a 32-deep serial fmac chain.
        float zsq = ((part[0] + part[1]) + (part[2] + part[3]))
                  + ((part[4] + part[5]) + (part[6] + part[7]));
        // Lanes l and l^16 hold disjoint K-halves of the same row: merge.
        zsq += __shfl_xor(zsq, 16, 32);   // now lane l holds ||z[row l15]||^2

        // ---- GEMM: 4 c-tiles x (K=64 as 2x K=32) ----
        // Separate accumulators per K-half: all 8 WMMAs are independent
        // (no D->C RAW chains); halves summed in the epilogue dual-issue adds.
        v8f acc0[4] = {v8f{}, v8f{}, v8f{}, v8f{}};
        v8f acc1[4] = {v8f{}, v8f{}, v8f{}, v8f{}};
        #pragma unroll
        for (int ct = 0; ct < 4; ++ct) {
            acc0[ct] = __builtin_amdgcn_wmma_f32_16x16x32_bf16(
                false, afrag[0], false, bfrag[ct][0], (short)0, acc0[ct], false, false);
            acc1[ct] = __builtin_amdgcn_wmma_f32_16x16x32_bf16(
                false, afrag[1], false, bfrag[ct][1], (short)0, acc1[ct], false, false);
        }

        // ---- Epilogue: affine transform + non-temporal store ----
        // Acc layout (32-bit C/D 16x16): lane n = column (l15), VGPR j = row
        // j + 8*hs. Row m's ||z||^2 lives in lane m.
        #pragma unroll
        for (int j = 0; j < 8; ++j) {
            const int m = j + hs * 8;
            const float zs = __shfl(zsq, m, 32);
            const float base = -0.5f * inv_var * zs;
            float* orow = op + (size_t)m * NC;
            #pragma unroll
            for (int ct = 0; ct < 4; ++ct) {
                const float dot = acc0[ct][j] + acc1[ct][j];
                __builtin_nontemporal_store(
                    fmaf(inv_var, dot, bias_c[ct] + base),
                    orow + ct * 16 + l15);
            }
        }
    }
}

extern "C" void kernel_launch(void* const* d_in, const int* in_sizes, int n_in,
                              void* d_out, int out_size, void* d_ws, size_t ws_size,
                              hipStream_t stream) {
    const float* z       = (const float*)d_in[0];
    const float* mu      = (const float*)d_in[1];
    const float* log_cov = (const float*)d_in[2];
    const float* prior   = (const float*)d_in[3];
    float* out = (float*)d_out;

    const int n_rows = in_sizes[0] / ND;      // N = 1048576
    const int ntiles = n_rows >> 4;           // 16-row tiles per wave-iteration

    int blocks = 2048;                        // 8 waves/block -> 16K waves
    if (blocks * 8 > ntiles) blocks = (ntiles + 7) / 8;
    if (blocks < 1) blocks = 1;

    lda_head_kernel<<<blocks, 256, 0, stream>>>(z, mu, log_cov, prior, out, n_rows);
}
